// StackedLSTM_61744449848184
// MI455X (gfx1250) — matrix operand order repaired and checked by reading, not compiled, not run on hardware
//
#include <hip/hip_runtime.h>

// Problem dims
#define B_ 64
#define T_ 256
#define D_ 128
#define H_ 1024
#define L_ 5
#define C_ 10
#define G_ 4096          // 4*H gate width
#define M_ (T_ * B_)     // 16384 rows for the big input GEMMs

#define NWG_R 256        // workgroups in recurrent kernel (4 hidden units each)
#define BARS_PER_LAYER (T_ + 1)

#if defined(__clang_major__) && (__clang_major__ < 23)
#define HAVE_TDM5 1      // ROCm 7.2 / clang-22: 5-arg tensor_load_to_lds builtin
#endif

typedef __attribute__((ext_vector_type(16))) __bf16 bf16x16;
typedef __attribute__((ext_vector_type(8)))  float  f32x8;
typedef __attribute__((ext_vector_type(4)))  unsigned u32x4;
typedef __attribute__((ext_vector_type(8)))  int      i32x8;
typedef __attribute__((ext_vector_type(4)))  int      i32x4;

union Frag { bf16x16 v; uint4 q[2]; };

__device__ __forceinline__ float sigm(float x) { return 1.0f / (1.0f + __expf(-x)); }

// ---------------------------------------------------------------------------
// CDNA5 async global->LDS copy (16B per lane), tracked by ASYNCcnt.
// LDS operand = low 32 bits of the generic pointer (flat-LDS aperture keeps
// the byte offset in addr[31:0]).
// ---------------------------------------------------------------------------
__device__ __forceinline__ void async_ld16(const void* g, void* l) {
    unsigned lds = (unsigned)(uintptr_t)l;
    unsigned long long ga = (unsigned long long)(uintptr_t)g;
    asm volatile("global_load_async_to_lds_b128 %0, %1, off"
                 :: "v"(lds), "v"(ga) : "memory");
}
__device__ __forceinline__ void wait_async0() {
    asm volatile("s_wait_asynccnt 0x0" ::: "memory");
}

// ---------------------------------------------------------------------------
// Grid barrier: monotonic counter (zeroed once per launch by zero_bar).
// ---------------------------------------------------------------------------
__device__ __forceinline__ void grid_barrier(int* bar, int target) {
    __syncthreads();
    if (threadIdx.x == 0) {
        __threadfence();
        atomicAdd(bar, 1);
        while (__hip_atomic_load(bar, __ATOMIC_RELAXED, __HIP_MEMORY_SCOPE_AGENT) < target) {
            __builtin_amdgcn_s_sleep(1);
        }
    }
    __syncthreads();
}

__global__ void zero_bar(int* bar) { if (threadIdx.x == 0) *bar = 0; }

// ---------------------------------------------------------------------------
// Pack x [B,T,D] fp32 -> X0 [(t*B+b), D] bf16 (time-major rows for the GEMM)
// ---------------------------------------------------------------------------
__global__ void pack_x(const float* __restrict__ x, __bf16* __restrict__ X0) {
    size_t i = (size_t)blockIdx.x * blockDim.x + threadIdx.x;
    if (i >= (size_t)M_ * D_) return;
    int d = (int)(i & (D_ - 1));
    size_t rb = i >> 7;            // row = t*B+b
    int b = (int)(rb & (B_ - 1));
    int t = (int)(rb >> 6);
    X0[i] = (__bf16)x[((size_t)b * T_ + t) * D_ + d];
}

// ---------------------------------------------------------------------------
// Transpose+convert: dst[n*K + k] = bf16(src[k*N + n])   (weights -> [N][K])
// ---------------------------------------------------------------------------
__global__ void tcvt(const float* __restrict__ src, __bf16* __restrict__ dst,
                     int K, int N) {
    size_t i = (size_t)blockIdx.x * blockDim.x + threadIdx.x;
    if (i >= (size_t)K * N) return;
    int n = (int)(i / K);
    int k = (int)(i % K);
    dst[i] = (__bf16)src[(size_t)k * N + n];
}

// ---------------------------------------------------------------------------
// Z = A @ B  with A [M][K] bf16 row-major, BT [N][K] bf16 (B transposed),
// out bf16 [M][N].  WG tile 128x64, 8 waves each computing a 32x32 patch
// (2x2 WMMA tiles).  K slices stream through double-buffered LDS via
// async-to-LDS loads: slice k+32 is in flight while WMMAs consume slice k.
// ---------------------------------------------------------------------------
__device__ __forceinline__ void stage_tiles(const __bf16* __restrict__ A,
                                            const __bf16* __restrict__ BT,
                                            __bf16* sA, __bf16* sB,
                                            int row0, int col0, int K, int k,
                                            int tid) {
    int r = tid >> 1, c = tid & 1;
    const __bf16* ga = A + (size_t)(row0 + r) * K + k + c * 16;
    __bf16* la = sA + r * 40 + c * 16;
    async_ld16(ga, la);
    async_ld16(ga + 8, la + 8);
    if (tid < 128) {
        const __bf16* gb = BT + (size_t)(col0 + r) * K + k + c * 16;
        __bf16* lb = sB + r * 40 + c * 16;
        async_ld16(gb, lb);
        async_ld16(gb + 8, lb + 8);
    }
}

__global__ __launch_bounds__(256) void gemm_z(const __bf16* __restrict__ A,
                                              const __bf16* __restrict__ BT,
                                              __bf16* __restrict__ Cout,
                                              int M, int N, int K) {
    __shared__ __attribute__((aligned(16))) __bf16 sA[2][128 * 40]; // pad stride 40
    __shared__ __attribute__((aligned(16))) __bf16 sB[2][64 * 40];

    int nBlkN = N / 64;
    int bm = blockIdx.x / nBlkN;
    int bn = blockIdx.x % nBlkN;
    int row0 = bm * 128, col0 = bn * 64;
    int tid = threadIdx.x, lane = tid & 31, wave = tid >> 5;
    int wm = wave & 3;   // 4 row-groups of 32
    int wn = wave >> 2;  // 2 col-groups of 32

    f32x8 acc[2][2] = {};

    int mrow_f = (lane & 15);
    int kbA = (lane < 16) ? 0 : 8;
    int kbB = (lane < 16) ? 0 : 8;

    // Prime the pipeline
    stage_tiles(A, BT, sA[0], sB[0], row0, col0, K, 0, tid);
    wait_async0();
    __syncthreads();

    int buf = 0;
    for (int k = 0; k < K; k += 32) {
        if (k + 32 < K)   // stage next slice into the other buffer (async)
            stage_tiles(A, BT, sA[buf ^ 1], sB[buf ^ 1], row0, col0, K, k + 32, tid);

        Frag a[2], b[2];
        for (int tm = 0; tm < 2; ++tm) {
            const uint4* p = (const uint4*)(sA[buf] + (wm * 32 + tm * 16 + mrow_f) * 40 + kbA);
            a[tm].q[0] = p[0]; a[tm].q[1] = p[2];          // K {kb..kb+7, kb+16..kb+23}
        }
        for (int tn = 0; tn < 2; ++tn) {
            const uint4* p = (const uint4*)(sB[buf] + (wn * 32 + tn * 16 + (lane & 15)) * 40 + kbB);
            b[tn].q[0] = p[0]; b[tn].q[1] = p[2];          // K {kb..kb+7, kb+16..kb+23}
        }
        for (int tm = 0; tm < 2; ++tm)
            for (int tn = 0; tn < 2; ++tn)
                acc[tm][tn] = __builtin_amdgcn_wmma_f32_16x16x32_bf16(
                    false, a[tm].v, false, b[tn].v, (short)0, acc[tm][tn], false, false);

        wait_async0();      // next buffer fully landed in LDS
        __syncthreads();    // ... for every wave
        buf ^= 1;
    }

    int hi8 = (lane >> 4) << 3;
    for (int tm = 0; tm < 2; ++tm)
        for (int tn = 0; tn < 2; ++tn)
            for (int r = 0; r < 8; ++r) {
                int row = row0 + wm * 32 + tm * 16 + r + hi8;
                int col = col0 + wn * 32 + tn * 16 + (lane & 15);
                Cout[(size_t)row * N + col] = (__bf16)acc[tm][tn][r];
            }
}

// ---------------------------------------------------------------------------
// Recurrent kernel: 256 WGs x 128 threads.  WG owns 4 hidden units = 16 gate
// columns; its 16x1024 bf16 Wh slice (32 KB) is pulled into LDS once via the
// Tensor Data Mover (3D tile: x=1024 K, y=4 units @ stride 1024 rows,
// z=4 gates @ stride 1024*1024 elems; TDM's linear LDS fill matches the
// lw[(gate*4+u)*1024 + k] layout), then reused for all 256 timesteps.
// Wave w computes the 16-row M-tile w of z (64x16), K=1024 from LDS weights
// and global h (L2-resident, double-buffered).  Gates regrouped via LDS for
// the elementwise update; grid barrier per timestep.
// ---------------------------------------------------------------------------
__global__ __launch_bounds__(128) void lstm_rec(const __bf16* __restrict__ WhT,  // [G][H]
                                                const __bf16* __restrict__ Z,    // [M][G]
                                                const float* __restrict__ bias,  // [G]
                                                __bf16* hbuf,                    // [2][B][H]
                                                __bf16* __restrict__ Hseq,       // [M][H]
                                                int* bar, int bar_base) {
    __shared__ __attribute__((aligned(16))) __bf16 lw[16 * 1024]; // [cc][k]
    __shared__ float lz[64 * 16];
    __shared__ float lc[64 * 4];

    int wg = blockIdx.x;
    int j0 = wg * 4;                     // hidden units j0..j0+3
    int tid = threadIdx.x, lane = tid & 31, wave = tid >> 5;

#ifdef HAVE_TDM5
    if (tid < 32) {   // one wave issues the TDM descriptor (TDM ignores EXEC)
        unsigned long long ga = (unsigned long long)(uintptr_t)(WhT + (size_t)j0 * H_);
        unsigned lds = (unsigned)(uintptr_t)lw;
        u32x4 g0 = { 1u,                                   // count=1 valid user D#
                     lds,                                  // lds_addr
                     (unsigned)(ga & 0xffffffffu),         // global_addr[31:0]
                     (unsigned)(((ga >> 32) & 0x01ffffffu) // global_addr[56:32]
                                | 0x80000000u) };          // type=2 ("image")
        i32x8 g1 = { (int)0x00010000,      // data_size=2B; no multicast/pad
                     (int)(1024u << 16),   // tensor_dim0=1024 (bits 79:48, low16)
                     (int)(1024u << 16),   // dim0 hi=0 | tensor_dim1=1024 low16
                     (int)(1024u << 16),   // dim1 hi=0 | tile_dim0=1024
                     (int)0x00040004,      // tile_dim1=4 | tile_dim2=4
                     (int)1024,            // tensor_dim0_stride=1024 (low 32)
                     (int)0,               // stride0 hi | stride1 low16 (1M -> 0)
                     (int)0x10 };          // tensor_dim1_stride[47:16] = 0x10 (1M)
        i32x4 g2 = { 4, 0, 0, 0 };         // tensor_dim2=4
        i32x4 g3 = { 0, 0, 0, 0 };
        __builtin_amdgcn_tensor_load_to_lds(g0, g1, g2, g3, 0);
        __builtin_amdgcn_s_wait_tensorcnt((short)0);
    }
#else
    // Fallback: per-lane async-to-LDS loads (2048 x 16B across 128 threads)
    for (int i = tid; i < 2048; i += 128) {
        int cc = i >> 7, ch = i & 127;
        int gate = cc >> 2, u = cc & 3;
        const __bf16* g = WhT + (size_t)(gate * H_ + j0 + u) * H_ + ch * 8;
        async_ld16(g, lw + cc * 1024 + ch * 8);
    }
    wait_async0();
#endif
    // Zero c and our slice of h(t=0)
    for (int i = tid; i < 256; i += 128) {
        lc[i] = 0.0f;
        int b = i >> 2, u = i & 3;
        hbuf[b * H_ + j0 + u] = (__bf16)0.0f;
    }
    __threadfence();
    grid_barrier(bar, (bar_base + 1) * NWG_R);

    int mrow = wave * 16 + (lane & 15);
    int kbA = (lane < 16) ? 0 : 8;
    int cc  = lane & 15;
    int kbB = (lane < 16) ? 0 : 8;

    for (int t = 0; t < T_; ++t) {
        const __bf16* hin = hbuf + (size_t)(t & 1) * (B_ * H_);
        __bf16* hout      = hbuf + (size_t)((t + 1) & 1) * (B_ * H_);

        f32x8 acc0 = {}, acc1 = {};                 // 2 interleaved accumulators
        for (int k = 0; k < H_; k += 64) {
            Frag a0, a1, b0, b1;
            const uint4* pa0 = (const uint4*)(hin + (size_t)mrow * H_ + k + kbA);
            a0.q[0] = pa0[0]; a0.q[1] = pa0[2];
            const uint4* pb0 = (const uint4*)(lw + cc * 1024 + k + kbB);
            b0.q[0] = pb0[0]; b0.q[1] = pb0[2];
            const uint4* pa1 = (const uint4*)(hin + (size_t)mrow * H_ + k + 32 + kbA);
            a1.q[0] = pa1[0]; a1.q[1] = pa1[2];
            const uint4* pb1 = (const uint4*)(lw + cc * 1024 + k + 32 + kbB);
            b1.q[0] = pb1[0]; b1.q[1] = pb1[2];
            acc0 = __builtin_amdgcn_wmma_f32_16x16x32_bf16(false, a0.v, false, b0.v,
                                                           (short)0, acc0, false, false);
            acc1 = __builtin_amdgcn_wmma_f32_16x16x32_bf16(false, a1.v, false, b1.v,
                                                           (short)0, acc1, false, false);
        }
        { // combine + park z tile in LDS for gate regrouping
            int rbase = wave * 16 + ((lane >> 4) << 3);
            int col = lane & 15;
            for (int r = 0; r < 8; ++r)
                lz[(rbase + r) * 16 + col] = acc0[r] + acc1[r];
        }
        __syncthreads();

        for (int i = tid; i < 256; i += 128) {       // 64 batch x 4 units
            int b = i >> 2, u = i & 3;
            int col = j0 + u;
            size_t zr = ((size_t)t * B_ + b) * G_;
            if (t + 1 < T_) __builtin_prefetch(Z + zr + (size_t)B_ * G_ + col, 0, 3);
            float vi = lz[b * 16 + 0 * 4 + u] + (float)Z[zr + 0 * H_ + col] + bias[0 * H_ + col];
            float vf = lz[b * 16 + 1 * 4 + u] + (float)Z[zr + 1 * H_ + col] + bias[1 * H_ + col];
            float vg = lz[b * 16 + 2 * 4 + u] + (float)Z[zr + 2 * H_ + col] + bias[2 * H_ + col];
            float vo = lz[b * 16 + 3 * 4 + u] + (float)Z[zr + 3 * H_ + col] + bias[3 * H_ + col];
            float c  = lc[i];
            float cn = sigm(vf) * c + sigm(vi) * tanhf(vg);
            float hn = sigm(vo) * tanhf(cn);
            lc[i] = cn;
            __bf16 hb = (__bf16)hn;
            hout[b * H_ + col] = hb;
            Hseq[((size_t)t * B_ + b) * H_ + col] = hb;
        }
        __threadfence();
        grid_barrier(bar, (bar_base + 2 + t) * NWG_R);
    }
}

// ---------------------------------------------------------------------------
// Head: out[b,c] = Hseq[last t, b] @ W_head + b_head   (tiny, scalar)
// ---------------------------------------------------------------------------
__global__ void head_k(const __bf16* __restrict__ Hseq, const float* __restrict__ W,
                       const float* __restrict__ bh, float* __restrict__ out) {
    int i = blockIdx.x * blockDim.x + threadIdx.x;
    if (i >= B_ * C_) return;
    int b = i / C_, c = i % C_;
    const __bf16* h = Hseq + ((size_t)(T_ - 1) * B_ + b) * H_;
    float s = bh[c];
    for (int k = 0; k < H_; ++k) s += (float)h[k] * W[k * C_ + c];
    out[i] = s;
}

// ---------------------------------------------------------------------------
// Workspace layout (bytes)
// ---------------------------------------------------------------------------
static constexpr size_t OFF_BAR  = 0;
static constexpr size_t OFF_X0   = 256;                                        // M*D bf16
static constexpr size_t OFF_WXT0 = OFF_X0   + (size_t)M_ * D_ * 2;             // G*D bf16
static constexpr size_t OFF_WXT  = OFF_WXT0 + (size_t)G_ * D_ * 2;             // 4 * G*H bf16
static constexpr size_t OFF_WHT  = OFF_WXT  + (size_t)4 * G_ * H_ * 2;         // 5 * G*H bf16
static constexpr size_t OFF_Z    = OFF_WHT  + (size_t)5 * G_ * H_ * 2;         // M*G bf16
static constexpr size_t OFF_HSEQ = OFF_Z    + (size_t)M_ * G_ * 2;             // M*H bf16
static constexpr size_t OFF_HBUF = OFF_HSEQ + (size_t)M_ * H_ * 2;             // 2*B*H bf16

extern "C" void kernel_launch(void* const* d_in, const int* in_sizes, int n_in,
                              void* d_out, int out_size, void* d_ws, size_t ws_size,
                              hipStream_t stream) {
    (void)in_sizes; (void)n_in; (void)out_size; (void)ws_size;

    const float* x      = (const float*)d_in[0];
    const float* Wx0    = (const float*)d_in[1];
    const float* Wxs    = (const float*)d_in[2];
    const float* Whs    = (const float*)d_in[3];
    const float* bs     = (const float*)d_in[4];
    const float* W_head = (const float*)d_in[5];
    const float* b_head = (const float*)d_in[6];
    float* out = (float*)d_out;

    char* ws = (char*)d_ws;
    int*    bar  = (int*)(ws + OFF_BAR);
    __bf16* X0   = (__bf16*)(ws + OFF_X0);
    __bf16* WxT0 = (__bf16*)(ws + OFF_WXT0);
    __bf16* WxT  = (__bf16*)(ws + OFF_WXT);
    __bf16* WhT  = (__bf16*)(ws + OFF_WHT);
    __bf16* Z    = (__bf16*)(ws + OFF_Z);
    __bf16* Hseq = (__bf16*)(ws + OFF_HSEQ);
    __bf16* hbuf = (__bf16*)(ws + OFF_HBUF);

    zero_bar<<<1, 32, 0, stream>>>(bar);
    pack_x<<<(M_ * D_ + 255) / 256, 256, 0, stream>>>(x, X0);
    tcvt<<<(D_ * G_ + 255) / 256, 256, 0, stream>>>(Wx0, WxT0, D_, G_);
    for (int l = 0; l < 4; ++l)
        tcvt<<<(H_ * G_ + 255) / 256, 256, 0, stream>>>(
            Wxs + (size_t)l * H_ * G_, WxT + (size_t)l * G_ * H_, H_, G_);
    for (int l = 0; l < 5; ++l)
        tcvt<<<(H_ * G_ + 255) / 256, 256, 0, stream>>>(
            Whs + (size_t)l * H_ * G_, WhT + (size_t)l * G_ * H_, H_, G_);

    const int gemm_grid = (M_ / 128) * (G_ / 64);   // 8192

    // Layer 0: Z = X0 @ Wx0 (K = D), then recurrence
    gemm_z<<<gemm_grid, 256, 0, stream>>>(X0, WxT0, Z, M_, G_, D_);
    lstm_rec<<<NWG_R, 128, 0, stream>>>(WhT, Z, bs, hbuf, Hseq, bar, 0 * BARS_PER_LAYER);

    // Layers 1..4: Z = Hseq(prev) @ Wx (K = H), then recurrence (Hseq reused)
    for (int l = 1; l < L_; ++l) {
        gemm_z<<<gemm_grid, 256, 0, stream>>>(Hseq, WxT + (size_t)(l - 1) * G_ * H_,
                                              Z, M_, G_, H_);
        lstm_rec<<<NWG_R, 128, 0, stream>>>(WhT + (size_t)l * G_ * H_, Z,
                                            bs + (size_t)l * G_, hbuf, Hseq, bar,
                                            l * BARS_PER_LAYER);
    }

    head_k<<<(B_ * C_ + 255) / 256, 256, 0, stream>>>(Hseq, W_head, b_head, out);
}
